// GNNClassifier_25666724561258
// MI455X (gfx1250) — matrix operand order repaired and checked by reading, not compiled, not run on hardware
//
#include <hip/hip_runtime.h>
#include <hip/hip_bf16.h>

// ---------------------------------------------------------------------------
// GNN classifier for MI455X (gfx1250, wave32).
//   L1: h1 = X W1;  agg1 = scatter(norm * h1[src]) + h1*dinv^2;  o1 = relu(agg1+b1)
//   L2: h2 = o1 W2 (bf16 WMMA, f32 acc); same aggregation; relu(+b2)
//   mean-pool per graph; logits = pooled @ Wc + bc; sigmoid.
// Edge phase is HBM/atomic bound (~3.5 GB); GEMM (3.3 GFLOP) uses
// v_wmma_f32_16x16x32_bf16.
// ---------------------------------------------------------------------------

typedef __attribute__((ext_vector_type(8)))  float  v8f;
typedef __attribute__((ext_vector_type(16))) __bf16 v16bf;
typedef __attribute__((ext_vector_type(8)))  __bf16 v8bf;

#define HID 128

// ---- zero init: deg[N], sums[G*128], cnt[G] -------------------------------
__global__ void __launch_bounds__(256) k_zero(float* __restrict__ deg,
                                              float* __restrict__ sums,
                                              float* __restrict__ cnt,
                                              int N, int G) {
  int t = blockIdx.x * 256 + threadIdx.x;
  if (t < N) deg[t] = 0.0f;
  if (t < G * HID) sums[t] = 0.0f;
  if (t < G) cnt[t] = 0.0f;
}

// ---- pack W2 (f32 [K=128][N=128] row-major) into WMMA B-operand layout ----
// Frag (ntile t, kblock kt): lane L holds column n = t*16 + (L&15),
// K = kt*32 + (L>>4)*8 + (e&7) + (e>>3)*16  for e = 0..15 (matches A layout).
__global__ void __launch_bounds__(256) k_pack_b(const float* __restrict__ W2,
                                                __bf16* __restrict__ Bp) {
  int p = blockIdx.x * 256 + threadIdx.x;
  if (p >= HID * HID) return;
  int e  = p & 15;
  int L  = (p >> 4) & 31;
  int kt = (p >> 9) & 3;
  int t  = p >> 11;
  int K  = kt * 32 + (L >> 4) * 8 + (e & 7) + (e >> 3) * 16;
  int n  = t * 16 + (L & 15);
  Bp[p] = (__bf16)W2[K * HID + n];
}

// ---- degree accumulation over dst ----------------------------------------
__global__ void __launch_bounds__(256) k_deg(const int* __restrict__ dst,
                                             float* __restrict__ deg, int E) {
  int e = blockIdx.x * 256 + threadIdx.x;
  if (e < E) atomicAdd(&deg[dst[e]], 1.0f);
}

// ---- deg -> dinv in place -------------------------------------------------
__global__ void __launch_bounds__(256) k_dinv(float* __restrict__ deg, int N) {
  int i = blockIdx.x * 256 + threadIdx.x;
  if (i < N) deg[i] = rsqrtf(deg[i] + 1.0f);
}

// ---- layer-1 transform (K=3) + self-loop init of agg ----------------------
__global__ void __launch_bounds__(256) k_l1(const float* __restrict__ x,
                                            const float* __restrict__ W1,
                                            const float* __restrict__ dinv,
                                            float* __restrict__ h,
                                            float* __restrict__ agg,
                                            int N) {
  long long t = (long long)blockIdx.x * 256 + threadIdx.x;
  if (t >= (long long)N * HID) return;
  int i = (int)(t >> 7);
  int j = (int)(t & (HID - 1));
  const float* xi = x + (long long)i * 3;
  float v = xi[0] * W1[j] + xi[1] * W1[HID + j] + xi[2] * W1[2 * HID + j];
  h[t] = v;
  float di = dinv[i];
  agg[t] = v * di * di;
}

// ---- edge scatter-gather: one wave per edge, float4 per lane --------------
__global__ void __launch_bounds__(256) k_edge(const int* __restrict__ src,
                                              const int* __restrict__ dst,
                                              const float* __restrict__ dinv,
                                              const float* __restrict__ h,
                                              float* __restrict__ agg,
                                              int E) {
  int gw = (int)(((long long)blockIdx.x * 256 + threadIdx.x) >> 5);
  if (gw >= E) return;
  int lane = threadIdx.x & 31;
  int s = src[gw];
  int d = dst[gw];
  float nrm = dinv[s] * dinv[d];
  const float4 m = *(const float4*)(h + (long long)s * HID + lane * 4);
  float* o = agg + (long long)d * HID + lane * 4;
  atomicAdd(o + 0, m.x * nrm);
  atomicAdd(o + 1, m.y * nrm);
  atomicAdd(o + 2, m.z * nrm);
  atomicAdd(o + 3, m.w * nrm);
}

// ---- post L1: relu(agg+b1) -> bf16 (feeds WMMA GEMM) ----------------------
__global__ void __launch_bounds__(256) k_post1(const float* __restrict__ agg,
                                               const float* __restrict__ b1,
                                               __bf16* __restrict__ ob,
                                               int N) {
  long long t = (long long)blockIdx.x * 256 + threadIdx.x;
  if (t >= (long long)N * HID) return;
  int j = (int)(t & (HID - 1));
  float v = fmaxf(agg[t] + b1[j], 0.0f);
  ob[t] = (__bf16)v;
}

// ---- layer-2 GEMM: [16 x 128] slab per block, 8 waves = 8 N-tiles ---------
// A layout (16-bit 16x32): lane = half*16 + row; elems 0..7 -> K = kb + half*8,
// elems 8..15 -> K = kb + 16 + half*8 (contiguous 16B per-lane loads).
__global__ void __launch_bounds__(256) k_gemm2(const __bf16* __restrict__ Ab,
                                               const __bf16* __restrict__ Bp,
                                               const float* __restrict__ dinv,
                                               float* __restrict__ h2,
                                               float* __restrict__ agg2,
                                               int N) {
  int wave = threadIdx.x >> 5;   // n-tile 0..7
  int lane = threadIdx.x & 31;
  int row  = lane & 15;
  int half = lane >> 4;
  int base = blockIdx.x * 16;

  int arow = base + row;
  if (arow >= N) arow = N - 1;  // clamp (safe duplicate load)
  const __bf16* Arow = Ab + (long long)arow * HID;

  v8f c = {};
#pragma unroll
  for (int kt = 0; kt < 4; ++kt) {
    union { v16bf v; v8bf h[2]; } a;
    const __bf16* p = Arow + kt * 32 + half * 8;
    a.h[0] = *(const v8bf*)(p);
    a.h[1] = *(const v8bf*)(p + 16);
    v16bf b = *(const v16bf*)(Bp + (((long long)wave * 4 + kt) * 32 + lane) * 16);
    c = __builtin_amdgcn_wmma_f32_16x16x32_bf16(false, a.v, false, b,
                                                (short)0, c, false, false);
  }

  int col = wave * 16 + (lane & 15);
#pragma unroll
  for (int v = 0; v < 8; ++v) {
    int node = base + v + 8 * half;      // C/D layout: lanes16-31 -> M = v+8
    if (node < N) {
      float val = c[v];
      long long idx = (long long)node * HID + col;
      h2[idx] = val;
      float di = dinv[node];
      agg2[idx] = val * di * di;
    }
  }
}

// ---- post L2: relu(agg+b2) + atomic mean-pool accumulation ----------------
__global__ void __launch_bounds__(256) k_post2_pool(const float* __restrict__ agg,
                                                    const float* __restrict__ b2,
                                                    const int* __restrict__ batch,
                                                    float* __restrict__ sums,
                                                    float* __restrict__ cnt,
                                                    int N) {
  long long t = (long long)blockIdx.x * 256 + threadIdx.x;
  if (t >= (long long)N * HID) return;
  int i = (int)(t >> 7);
  int j = (int)(t & (HID - 1));
  float v = fmaxf(agg[t] + b2[j], 0.0f);
  int g = batch[i];
  atomicAdd(&sums[(long long)g * HID + j], v);
  if (j == 0) atomicAdd(&cnt[g], 1.0f);
}

// ---- head: one block per graph, LDS tree reduction, sigmoid ---------------
__global__ void __launch_bounds__(128) k_head(const float* __restrict__ sums,
                                              const float* __restrict__ cnt,
                                              const float* __restrict__ Wc,
                                              const float* __restrict__ bc,
                                              float* __restrict__ out, int G) {
  __shared__ float red[HID];
  int g = blockIdx.x;
  int t = threadIdx.x;
  float c = fmaxf(cnt[g], 1.0f);
  red[t] = (sums[(long long)g * HID + t] / c) * Wc[t];
  __syncthreads();
#pragma unroll
  for (int s = 64; s > 0; s >>= 1) {
    if (t < s) red[t] += red[t + s];
    __syncthreads();
  }
  if (t == 0) {
    float z = red[0] + bc[0];
    out[g] = 1.0f / (1.0f + __expf(-z));
  }
}

// ---------------------------------------------------------------------------
extern "C" void kernel_launch(void* const* d_in, const int* in_sizes, int n_in,
                              void* d_out, int out_size, void* d_ws, size_t ws_size,
                              hipStream_t stream) {
  const float* x  = (const float*)d_in[0];
  const int*   ei = (const int*)d_in[1];
  const int*   batch = (const int*)d_in[2];
  const float* W1 = (const float*)d_in[3];
  const float* b1 = (const float*)d_in[4];
  const float* W2 = (const float*)d_in[5];
  const float* b2 = (const float*)d_in[6];
  const float* Wc = (const float*)d_in[7];
  const float* bc = (const float*)d_in[8];
  float* out = (float*)d_out;

  const int N = in_sizes[0] / 3;
  const int E = in_sizes[1] / 2;
  const int G = out_size;
  const int* src = ei;
  const int* dst = ei + E;

  // -------- workspace carve-up (256B aligned) --------
  char* w = (char*)d_ws;
  size_t off = 0;
  auto carve = [&](size_t bytes) -> void* {
    void* p = (void*)(w + off);
    off = (off + bytes + 255) & ~(size_t)255;
    return p;
  };
  float*  dinv  = (float*)carve((size_t)N * 4);
  float*  hbuf  = (float*)carve((size_t)N * HID * 4);   // h1, reused as h2
  float*  abuf  = (float*)carve((size_t)N * HID * 4);   // agg1, reused as agg2
  __bf16* o1b   = (__bf16*)carve((size_t)N * HID * 2);  // relu(L1) in bf16
  __bf16* Bp    = (__bf16*)carve((size_t)HID * HID * 2);
  float*  sums  = (float*)carve((size_t)G * HID * 4);
  float*  cnt   = (float*)carve((size_t)G * 4);

  const long long NH = (long long)N * HID;
  const int nhBlocks   = (int)((NH + 255) / 256);
  const int edgeThr    = (int)(((long long)E * 32 + 255) / 256);
  const int zeroBlocks = (int)((((long long)(N > G * HID ? N : G * HID)) + 255) / 256);

  // 0. zero deg / pooled sums / counts (deterministic every launch)
  k_zero<<<zeroBlocks, 256, 0, stream>>>(dinv, sums, cnt, N, G);
  // 1. pack W2 into WMMA B-operand bf16 layout
  k_pack_b<<<(HID * HID + 255) / 256, 256, 0, stream>>>(W2, Bp);
  // 2. degrees (over dst) -> dinv
  k_deg<<<(E + 255) / 256, 256, 0, stream>>>(dst, dinv, E);
  k_dinv<<<(N + 255) / 256, 256, 0, stream>>>(dinv, N);
  // 3. layer 1: transform + self-loop init, edge scatter, relu->bf16
  k_l1<<<nhBlocks, 256, 0, stream>>>(x, W1, dinv, hbuf, abuf, N);
  k_edge<<<edgeThr, 256, 0, stream>>>(src, dst, dinv, hbuf, abuf, E);
  k_post1<<<nhBlocks, 256, 0, stream>>>(abuf, b1, o1b, N);
  // 4. layer 2: WMMA GEMM (writes h2 + self-loop init), edge scatter
  k_gemm2<<<(N + 15) / 16, 256, 0, stream>>>(o1b, Bp, dinv, hbuf, abuf, N);
  k_edge<<<edgeThr, 256, 0, stream>>>(src, dst, dinv, hbuf, abuf, E);
  // 5. relu + mean-pool accumulation, then per-graph head
  k_post2_pool<<<nhBlocks, 256, 0, stream>>>(abuf, b2, batch, sums, cnt, N);
  k_head<<<G, HID, 0, stream>>>(sums, cnt, Wc, bc, out, G);
}
